// MultiHeadAttention_6021544149499
// MI455X (gfx1250) — matrix operand order repaired and checked by reading, not compiled, not run on hardware
//
#include <hip/hip_runtime.h>

// ---------------------------------------------------------------------------
// MultiHeadAttention forward for MI455X (gfx1250, wave32, WMMA bf16)
// Pipeline: f32->bf16 convert -> 3x WMMA GEMM (QKV proj, head-split out)
//           -> causal flash-attention (WMMA QK^T + PV, online softmax)
//           -> WMMA GEMM output projection (f32 out + bias)
// Data movement: GLOBAL_LOAD_ASYNC_TO_LDS_B128 (ASYNCcnt) when available,
// double-buffered LDS tiles; falls back to sync VGPR-mediated copies.
// ---------------------------------------------------------------------------

typedef unsigned int  u32;
typedef unsigned short u16;
typedef __attribute__((ext_vector_type(4)))  u32    u32x4;
typedef __attribute__((ext_vector_type(4)))  int    i32x4;
typedef __attribute__((ext_vector_type(4)))  float  f32x4;
typedef __attribute__((ext_vector_type(16))) __bf16 v16bf;
typedef __attribute__((ext_vector_type(8)))  float  v8f;

#define D_MODEL 1024
#define NHEADS  16
#define DEPTH   64
#define BATCH   2
#define SEQ     2048
#define MROWS   (BATCH * SEQ)   // 4096

#define AS1 __attribute__((address_space(1)))
#define AS3 __attribute__((address_space(3)))

__device__ __forceinline__ u16 f2bf(float f) {
    union { float f; u32 u; } v; v.f = f;
    return (u16)((v.u + 0x7FFFu + ((v.u >> 16) & 1u)) >> 16);  // RNE
}

// ---- async global->LDS copy (16B per lane), with sync fallback -------------
#if __has_builtin(__builtin_amdgcn_global_load_async_to_lds_b128)
#define HAVE_ASYNC_LDS 1
__device__ __forceinline__ void g2l_b128(const u16* g, u16* l) {
    __builtin_amdgcn_global_load_async_to_lds_b128(
        (AS1 i32x4*)g, (AS3 i32x4*)l, 0, 0);
}
#else
#define HAVE_ASYNC_LDS 0
__device__ __forceinline__ void g2l_b128(const u16* g, u16* l) {
    *(u32x4*)l = *(const u32x4*)g;
}
#endif
// Wait until ASYNCcnt <= N. Harmless no-op in the fallback path (cnt==0).
__device__ __forceinline__ void wait_async_all() {
    asm volatile("s_wait_asynccnt 0x0" ::: "memory");
}
__device__ __forceinline__ void wait_async_le8() {
    asm volatile("s_wait_asynccnt 0x8" ::: "memory");
}

union FragU { v16bf bf; u32x4 q[2]; };

// A-operand fragment (16x32 bf16): lane m = lane&15, kh = lane>>4.
// VGPR0-3 <- K=[8kh,8kh+8), VGPR4-7 <- K=[16+8kh,16+8kh+8)  (ISA 7.12.2)
__device__ __forceinline__ v16bf frag_a(const u16* row, int kh) {
    FragU f;
    f.q[0] = *(const u32x4*)(row + 8 * kh);
    f.q[1] = *(const u32x4*)(row + 16 + 8 * kh);
    return f.bf;
}
// B-operand fragment (32x16 bf16): lane n = lane&15, kh = lane>>4.
// VGPR0-3 <- K=[8kh,8kh+8), VGPR4-7 <- K=[16+8kh,16+8kh+8) of column n (row pointer = column n of B^T)
__device__ __forceinline__ v16bf frag_b(const u16* row, int kh) {
    FragU f;
    f.q[0] = *(const u32x4*)(row + 8 * kh);
    f.q[1] = *(const u32x4*)(row + 16 + 8 * kh);
    return f.bf;
}

__device__ __forceinline__ v8f wmma_bf16(v16bf a, v16bf b, v8f c) {
    return __builtin_amdgcn_wmma_f32_16x16x32_bf16(
        /*neg_a=*/false, a, /*neg_b=*/false, b,
        /*c_mod=*/(short)0, c, /*reuse_a=*/false, /*reuse_b=*/false);
}

// ---------------------------------------------------------------------------
// f32 -> bf16 elementwise convert (4 elems / thread)
// ---------------------------------------------------------------------------
__global__ __launch_bounds__(256)
void cvt_bf16_kernel(const float* __restrict__ in, u16* __restrict__ out, int n4) {
    int i = blockIdx.x * 256 + threadIdx.x;
    if (i >= n4) return;
    f32x4 v = ((const f32x4*)in)[i];
    u32 lo = (u32)f2bf(v.x) | ((u32)f2bf(v.y) << 16);
    u32 hi = (u32)f2bf(v.z) | ((u32)f2bf(v.w) << 16);
    ((u32*)out)[2 * i + 0] = lo;
    ((u32*)out)[2 * i + 1] = hi;
}

// f32 weight [K][N] -> bf16 transposed [N][K]
__global__ __launch_bounds__(256)
void cvt_w_transpose_kernel(const float* __restrict__ in, u16* __restrict__ out) {
    int idx = blockIdx.x * 256 + threadIdx.x;        // over 1M, out-linear
    int n = idx >> 10, k = idx & 1023;
    out[idx] = f2bf(in[(size_t)k * D_MODEL + n]);
}

// ---------------------------------------------------------------------------
// bf16 GEMM: Y[M=4096][N=1024] = A[M][1024] * BT[N][1024]^T + bias
// mode 0: write bf16 head-split [B,H,S,64];  mode 1: write f32 row-major
// Block 128x128x64, 256 thr = 8 waves, wave tile 32x64 (16 WMMA / k-tile).
// Double-buffered LDS, async global->LDS loads.
// ---------------------------------------------------------------------------
#define GBM 128
#define GBN 128
#define GBK 64
#define TPAD 72   // row pad (u16) -> 144B rows, 16B aligned, conflict-spread

__global__ __launch_bounds__(256)
void gemm_bf16_kernel(const u16* __restrict__ A, const u16* __restrict__ BT,
                      const float* __restrict__ bias,
                      u16* __restrict__ outH, float* __restrict__ outF, int mode) {
    __shared__ __align__(16) u16 At[2][GBM * TPAD];
    __shared__ __align__(16) u16 Bt[2][GBN * TPAD];

    const int Kdim = D_MODEL;
    int tid = threadIdx.x, lane = tid & 31, wid = tid >> 5;
    int mlane = lane & 15, kh = lane >> 4;
    int m0g = blockIdx.y * GBM, n0g = blockIdx.x * GBN;
    int wm = (wid >> 1) * 32, wn = (wid & 1) * 64;

    const v8f vzero = {0.f, 0.f, 0.f, 0.f, 0.f, 0.f, 0.f, 0.f};
    v8f acc[2][4];
#pragma unroll
    for (int i = 0; i < 2; i++)
#pragma unroll
        for (int j = 0; j < 4; j++) acc[i][j] = vzero;

    int lr = tid >> 1, lc = (tid & 1) * 32;   // 2 threads/row, 32 u16 each
    const u16* agbase = A  + (size_t)(m0g + lr) * Kdim + lc;
    const u16* bgbase = BT + (size_t)(n0g + lr) * Kdim + lc;

    auto issue_tile = [&](int k0, int buf) {
        const u16* ag = agbase + k0;
        const u16* bg = bgbase + k0;
        u16* al = &At[buf][lr * TPAD + lc];
        u16* bl = &Bt[buf][lr * TPAD + lc];
#pragma unroll
        for (int i = 0; i < 4; i++) {   // 8 async b128 per thread per k-tile
            g2l_b128(ag + 8 * i, al + 8 * i);
            g2l_b128(bg + 8 * i, bl + 8 * i);
        }
    };

    const int T = Kdim / GBK;   // 16 k-tiles
    issue_tile(0, 0);
    for (int t = 0; t < T; ++t) {
        if (t + 1 < T) {
            issue_tile((t + 1) * GBK, (t + 1) & 1);
            wait_async_le8();               // tile t complete (in-order done)
        } else {
            wait_async_all();
        }
        if (t + 2 < T) {                    // warm L2 for tile t+2
            __builtin_prefetch(agbase + (t + 2) * GBK, 0, 3);
            __builtin_prefetch(bgbase + (t + 2) * GBK, 0, 3);
        }
        __syncthreads();

        const u16* Ab = At[t & 1];
        const u16* Bb = Bt[t & 1];
#pragma unroll
        for (int ks2 = 0; ks2 < 2; ++ks2) {
            v16bf af[2], bfr[4];
#pragma unroll
            for (int i = 0; i < 2; i++)
                af[i] = frag_a(&Ab[(wm + i * 16 + mlane) * TPAD + ks2 * 32], kh);
#pragma unroll
            for (int j = 0; j < 4; j++)
                bfr[j] = frag_b(&Bb[(wn + j * 16 + mlane) * TPAD + ks2 * 32], kh);
#pragma unroll
            for (int i = 0; i < 2; i++)
#pragma unroll
                for (int j = 0; j < 4; j++)
                    acc[i][j] = wmma_bf16(af[i], bfr[j], acc[i][j]);
        }
        __syncthreads();                    // readers done before buf reused
    }

    // Epilogue. C layout: lanes 0-15: M=vgpr; lanes 16-31: M=vgpr+8; N=lane&15.
#pragma unroll
    for (int i = 0; i < 2; i++) {
#pragma unroll
        for (int j = 0; j < 4; j++) {
            int gn = n0g + wn + j * 16 + mlane;
            float bv = bias[gn];
#pragma unroll
            for (int r = 0; r < 8; r++) {
                int gm = m0g + wm + i * 16 + kh * 8 + r;
                float val = acc[i][j][r] + bv;
                if (mode == 0) {
                    int b = gm >> 11, s = gm & 2047;
                    int h = gn >> 6,  d = gn & 63;
                    size_t idx = ((((size_t)b * NHEADS + h) * SEQ) + s) * DEPTH + d;
                    outH[idx] = f2bf(val);
                } else {
                    outF[(size_t)gm * D_MODEL + gn] = val;
                }
            }
        }
    }
}

// ---------------------------------------------------------------------------
// Causal flash attention: one block per (b*H+h, 64-row q tile), 128 thr = 4 waves.
// Each wave owns 16 q rows. S_tile = Q*K^T (WMMA), online softmax in fp32,
// O += P*V (WMMA), final scale by 1/l. Q/K tiles via async->LDS.
// ---------------------------------------------------------------------------
#define QT   64
#define KT   64
#define HPAD 72   // 144B rows, 16B aligned
#define SPAD 68

__global__ __launch_bounds__(128)
void flash_attn_kernel(const u16* __restrict__ Qp, const u16* __restrict__ Kp,
                       const u16* __restrict__ Vp, u16* __restrict__ Op) {
    __shared__ __align__(16) u16   Qt_s[QT * HPAD];
    __shared__ __align__(16) u16   Kt_s[KT * HPAD];
    __shared__ __align__(16) u16   Vt_s[DEPTH * HPAD];  // transposed: [d][key]
    __shared__ __align__(16) u16   Pt_s[QT * HPAD];     // P (bf16 probs)
    __shared__ __align__(16) float Sp_s[QT * SPAD];     // logits fp32
    __shared__ float corrL[QT];
    __shared__ float invL[QT];

    int tid = threadIdx.x, lane = tid & 31, wid = tid >> 5;
    int mlane = lane & 15, kh = lane >> 4;
    int qt = blockIdx.x;          // q tile index (0..31)
    int bh = blockIdx.y;          // b*NHEADS + h
    int b = bh >> 4, h = bh & 15;
    size_t headoff = (size_t)bh * SEQ * DEPTH;
    int q0 = qt * QT;
    int lrr = tid >> 1, lcc = (tid & 1) * 32;  // tile-load assignment

    // Q tile [64 x 64] async into LDS (first in-loop wait covers it)
    {
        const u16* src = Qp + headoff + (size_t)(q0 + lrr) * DEPTH + lcc;
        u16* dst = &Qt_s[lrr * HPAD + lcc];
#pragma unroll
        for (int i = 0; i < 4; i++) g2l_b128(src + 8 * i, dst + 8 * i);
    }

    float row_m = -3.0e38f, row_l = 0.f;   // valid in threads 0..63 (row owners)
    const v8f vzero = {0.f, 0.f, 0.f, 0.f, 0.f, 0.f, 0.f, 0.f};
    v8f o[4];
#pragma unroll
    for (int nt = 0; nt < 4; nt++) o[nt] = vzero;

    for (int j = 0; j <= qt; ++j) {        // causal: only lower-triangular tiles
        int k0 = j * KT;
        __syncthreads();                    // prev-iter readers done w/ K/V/P
        // K tile async; V tile through VGPRs with transpose into [d][key]
        {
            const u16* ks = Kp + headoff + (size_t)(k0 + lrr) * DEPTH + lcc;
            u16* kd = &Kt_s[lrr * HPAD + lcc];
#pragma unroll
            for (int i = 0; i < 4; i++) g2l_b128(ks + 8 * i, kd + 8 * i);

            const u16* vs = Vp + headoff + (size_t)(k0 + lrr) * DEPTH + lcc;
            u16 tmp[32];
            *(u32x4*)(tmp + 0)  = ((const u32x4*)vs)[0];
            *(u32x4*)(tmp + 8)  = ((const u32x4*)vs)[1];
            *(u32x4*)(tmp + 16) = ((const u32x4*)vs)[2];
            *(u32x4*)(tmp + 24) = ((const u32x4*)vs)[3];
#pragma unroll
            for (int i = 0; i < 32; i++) Vt_s[(lcc + i) * HPAD + lrr] = tmp[i];
        }
        wait_async_all();
        __syncthreads();

        // S = scale * Q K^T   (wave's 16 rows x 64 keys; k over d=64 in 2 steps)
        v8f s[4];
#pragma unroll
        for (int nt = 0; nt < 4; nt++) s[nt] = vzero;
#pragma unroll
        for (int ks2 = 0; ks2 < 2; ++ks2) {
            v16bf a = frag_a(&Qt_s[(wid * 16 + mlane) * HPAD + ks2 * 32], kh);
#pragma unroll
            for (int nt = 0; nt < 4; nt++) {
                v16bf bb = frag_b(&Kt_s[(nt * 16 + mlane) * HPAD + ks2 * 32], kh);
                s[nt] = wmma_bf16(a, bb, s[nt]);
            }
        }
        bool diag = (j == qt);
#pragma unroll
        for (int nt = 0; nt < 4; nt++) {
#pragma unroll
            for (int r = 0; r < 8; r++) {
                int m = wid * 16 + kh * 8 + r;
                int n = nt * 16 + mlane;
                float v = s[nt][r] * 0.125f;          // 1/sqrt(64)
                if (diag && n > m) v = -1e9f;          // causal NEG
                Sp_s[m * SPAD + n] = v;
            }
        }
        __syncthreads();

        // Online softmax: thread t<64 owns row t (state stays in its registers)
        if (tid < QT) {
            const float* srow = &Sp_s[tid * SPAD];
            float mx = row_m;
#pragma unroll 8
            for (int n = 0; n < KT; n++) mx = fmaxf(mx, srow[n]);
            float corr = __expf(row_m - mx);
            u16* prow = &Pt_s[tid * HPAD];
            float sum = 0.f;
#pragma unroll 8
            for (int n = 0; n < KT; n++) {
                float p = __expf(srow[n] - mx);
                prow[n] = f2bf(p);
                sum += p;
            }
            row_l = row_l * corr + sum;
            row_m = mx;
            corrL[tid] = corr;
        }
        __syncthreads();

        // Rescale O, then O += P * V
        float cf[8];
#pragma unroll
        for (int r = 0; r < 8; r++) cf[r] = corrL[wid * 16 + kh * 8 + r];
#pragma unroll
        for (int nt = 0; nt < 4; nt++)
#pragma unroll
            for (int r = 0; r < 8; r++) o[nt][r] *= cf[r];
#pragma unroll
        for (int ks2 = 0; ks2 < 2; ++ks2) {
            v16bf a = frag_a(&Pt_s[(wid * 16 + mlane) * HPAD + ks2 * 32], kh);
#pragma unroll
            for (int nt = 0; nt < 4; nt++) {
                v16bf bb = frag_b(&Vt_s[(nt * 16 + mlane) * HPAD + ks2 * 32], kh);
                o[nt] = wmma_bf16(a, bb, o[nt]);
            }
        }
    }

    __syncthreads();
    if (tid < QT) invL[tid] = 1.0f / row_l;
    __syncthreads();

    float il[8];
#pragma unroll
    for (int r = 0; r < 8; r++) il[r] = invL[wid * 16 + kh * 8 + r];
#pragma unroll
    for (int nt = 0; nt < 4; nt++) {
#pragma unroll
        for (int r = 0; r < 8; r++) {
            int m = wid * 16 + kh * 8 + r;
            int n = nt * 16 + mlane;
            size_t idx = ((size_t)b * SEQ + (q0 + m)) * D_MODEL + h * DEPTH + n;
            Op[idx] = f2bf(o[nt][r] * il[r]);
        }
    }
}

// ---------------------------------------------------------------------------
// Host-side launcher
// ---------------------------------------------------------------------------
extern "C" void kernel_launch(void* const* d_in, const int* in_sizes, int n_in,
                              void* d_out, int out_size, void* d_ws, size_t ws_size,
                              hipStream_t stream) {
    (void)in_sizes; (void)n_in; (void)out_size; (void)ws_size;
    const float* q  = (const float*)d_in[0];
    const float* k  = (const float*)d_in[1];
    const float* v  = (const float*)d_in[2];
    // d_in[3]: causal mask — applied analytically in flash_attn_kernel
    const float* wq = (const float*)d_in[4];
    const float* bq = (const float*)d_in[5];
    const float* wk = (const float*)d_in[6];
    const float* bk = (const float*)d_in[7];
    const float* wv = (const float*)d_in[8];
    const float* bv = (const float*)d_in[9];
    const float* wo = (const float*)d_in[10];
    const float* bo = (const float*)d_in[11];

    char* ws = (char*)d_ws;
    const size_t XSZ = (size_t)MROWS * D_MODEL * sizeof(u16);    // 8 MiB
    const size_t WSZ = (size_t)D_MODEL * D_MODEL * sizeof(u16);  // 2 MiB
    u16* Xq  = (u16*)(ws + 0 * XSZ);
    u16* Xk  = (u16*)(ws + 1 * XSZ);
    u16* Xv  = (u16*)(ws + 2 * XSZ);
    u16* Qh  = (u16*)(ws + 3 * XSZ);
    u16* Kh  = (u16*)(ws + 4 * XSZ);
    u16* Vh  = (u16*)(ws + 5 * XSZ);
    u16* Ao  = (u16*)(ws + 6 * XSZ);
    u16* WqT = (u16*)(ws + 7 * XSZ + 0 * WSZ);
    u16* WkT = (u16*)(ws + 7 * XSZ + 1 * WSZ);
    u16* WvT = (u16*)(ws + 7 * XSZ + 2 * WSZ);
    u16* WoT = (u16*)(ws + 7 * XSZ + 3 * WSZ);

    // 1) Precision convert
    int n4 = MROWS * D_MODEL / 4;
    dim3 cg((n4 + 255) / 256);
    cvt_bf16_kernel<<<cg, 256, 0, stream>>>(q, Xq, n4);
    cvt_bf16_kernel<<<cg, 256, 0, stream>>>(k, Xk, n4);
    cvt_bf16_kernel<<<cg, 256, 0, stream>>>(v, Xv, n4);
    dim3 wg((D_MODEL * D_MODEL + 255) / 256);
    cvt_w_transpose_kernel<<<wg, 256, 0, stream>>>(wq, WqT);
    cvt_w_transpose_kernel<<<wg, 256, 0, stream>>>(wk, WkT);
    cvt_w_transpose_kernel<<<wg, 256, 0, stream>>>(wv, WvT);
    cvt_w_transpose_kernel<<<wg, 256, 0, stream>>>(wo, WoT);

    // 2) QKV projections (head-split bf16 out)
    dim3 gg(D_MODEL / GBN, MROWS / GBM);   // (8, 32)
    gemm_bf16_kernel<<<gg, 256, 0, stream>>>(Xq, WqT, bq, Qh, (float*)nullptr, 0);
    gemm_bf16_kernel<<<gg, 256, 0, stream>>>(Xk, WkT, bk, Kh, (float*)nullptr, 0);
    gemm_bf16_kernel<<<gg, 256, 0, stream>>>(Xv, WvT, bv, Vh, (float*)nullptr, 0);

    // 3) Causal flash attention -> Ao [B,S,D_MODEL] bf16
    dim3 ag(SEQ / QT, BATCH * NHEADS);     // (32, 32)
    flash_attn_kernel<<<ag, 128, 0, stream>>>(Qh, Kh, Vh, Ao);

    // 4) Output projection -> d_out f32
    gemm_bf16_kernel<<<gg, 256, 0, stream>>>(Ao, WoT, bo, (u16*)nullptr, (float*)d_out, 1);
}
